// SelectiveSSM_25855703121984
// MI455X (gfx1250) — matrix operand order repaired and checked by reading, not compiled, not run on hardware
//
#include <hip/hip_runtime.h>
#include <hip/hip_bf16.h>

// Problem constants (from reference)
#define B_SZ 2
#define L_SZ 1024
#define DM   1024          // D_MODEL
#define DI   2048          // D_INNER
#define DS   16            // D_STATE
#define DC   4             // D_CONV
#define DTR  64            // DT_RANK
#define NTOK (B_SZ * L_SZ) // 2048 tokens
#define NCH  8             // scan chunks
#define CHL  (L_SZ / NCH)  // 128 steps per chunk

typedef unsigned short ushort_t;
typedef __attribute__((ext_vector_type(16))) __bf16       v16bf;
typedef __attribute__((ext_vector_type(8)))  float        v8f;
typedef __attribute__((ext_vector_type(4)))  unsigned int u32x4;

// ---------- helpers ----------
__device__ __forceinline__ ushort_t f2bf(float x) {
  unsigned int u = __builtin_bit_cast(unsigned int, x);
  u = u + 0x7FFFu + ((u >> 16) & 1u);   // round-to-nearest-even
  return (ushort_t)(u >> 16);
}

__device__ __forceinline__ v16bf load_frag16(const ushort_t* p0, const ushort_t* p1) {
  union { u32x4 q[2]; v16bf v; } f;
  f.q[0] = *reinterpret_cast<const u32x4*>(p0);   // lane elements 0..7
  f.q[1] = *reinterpret_cast<const u32x4*>(p1);   // lane elements 8..15
  return f.v;
}

__device__ __forceinline__ float sigmoidf_(float x) { return 1.0f / (1.0f + expf(-x)); }

// ---------- WMMA GEMM, register-blocked: each wave computes a 16 x (16*NBLK) strip ----------
// C[M,N] = A[M,K](bf16 row-major, lda) * B  where B is stored transposed bf16 [N,K] (ldbt)
// MODE 0: C = acc ; MODE 1: C = softplus(acc + bias[n])
template <int NBLK, int MODE>
__global__ void wmma_gemm_bf16_kernel(const ushort_t* __restrict__ A, int lda,
                                      const ushort_t* __restrict__ BT, int ldbt,
                                      float* __restrict__ C, int ldc,
                                      int M, int N, int K,
                                      const float* __restrict__ bias) {
  const int gtid    = (int)(blockIdx.x * blockDim.x + threadIdx.x);
  const int wave    = gtid >> 5;
  const int lane    = (int)(threadIdx.x & 31);
  const int groupsN = N / (16 * NBLK);
  const int tm = wave / groupsN;
  const int tg = wave - tm * groupsN;
  if (tm >= (M >> 4)) return;               // wave-uniform: EXEC stays all-ones for WMMA

  const int half = lane >> 4;               // 0: lanes 0-15, 1: lanes 16-31
  const int l15  = lane & 15;

  const ushort_t* arow = A + (size_t)(tm * 16 + l15) * lda;
  const ushort_t* brow[NBLK];
#pragma unroll
  for (int i = 0; i < NBLK; ++i)
    brow[i] = BT + (size_t)(tg * NBLK * 16 + i * 16 + l15) * ldbt;

  v8f acc[NBLK];
#pragma unroll
  for (int i = 0; i < NBLK; ++i) acc[i] = v8f{};

  for (int k = 0; k < K; k += 32) {
    // A: elems 0..7 -> K = half*8 .. +7 ; elems 8..15 -> K = 16+half*8 .. +7
    const v16bf a = load_frag16(arow + k + half * 8, arow + k + 16 + half * 8);
#pragma unroll
    for (int i = 0; i < NBLK; ++i) {
      // B: elems 0..7 -> K = half*8 .. +7 ; elems 8..15 -> K = 16+half*8 .. +7
      const v16bf b = load_frag16(brow[i] + k + half * 8, brow[i] + k + 16 + half * 8);
      acc[i] = __builtin_amdgcn_wmma_f32_16x16x32_bf16(false, a, false, b,
                                                       (short)0, acc[i], false, false);
    }
  }

#pragma unroll
  for (int i = 0; i < NBLK; ++i) {
    const int col = (tg * NBLK + i) * 16 + l15;
    const float bv = (MODE == 1) ? bias[col] : 0.0f;
#pragma unroll
    for (int r = 0; r < 8; ++r) {
      const int row = tm * 16 + half * 8 + r;  // C/D layout: VGPR r -> M = half*8 + r
      float v = acc[i][r];
      if (MODE == 1) {
        v += bv;
        v = (v > 20.0f) ? v : log1pf(expf(v)); // softplus
      }
      C[(size_t)row * ldc + col] = v;
    }
  }
}

// ---------- f32 -> bf16 transpose via LDS tile (weights, once per call) ----------
__global__ void transpose_f32_to_bf16(const float* __restrict__ in, ushort_t* __restrict__ out,
                                      int rows, int cols) {
  __shared__ ushort_t tile[32][33];
  const int c  = (int)(blockIdx.x * 32 + threadIdx.x);
  const int r0 = (int)(blockIdx.y * 32);
  for (int i = (int)threadIdx.y; i < 32; i += (int)blockDim.y) {
    const int r = r0 + i;
    ushort_t v = 0;
    if (r < rows && c < cols) v = f2bf(in[(size_t)r * cols + c]);
    tile[i][threadIdx.x] = v;
  }
  __syncthreads();
  const int oc = r0 + (int)threadIdx.x;             // out col = in row
  for (int i = (int)threadIdx.y; i < 32; i += (int)blockDim.y) {
    const int orow = (int)(blockIdx.x * 32) + i;    // out row = in col
    if (orow < cols && oc < rows)
      out[(size_t)orow * rows + oc] = tile[threadIdx.x][i];
  }
}

// ---------- f32 -> bf16 dense convert (with source row stride) ----------
__global__ void convert_bf16_kernel(const float* __restrict__ in, int srcStride,
                                    ushort_t* __restrict__ out, int rows, int cols) {
  const size_t n = (size_t)rows * cols;
  for (size_t i = (size_t)blockIdx.x * blockDim.x + threadIdx.x; i < n;
       i += (size_t)gridDim.x * blockDim.x) {
    const size_t r = i / (size_t)cols;
    const size_t c = i - r * (size_t)cols;
    out[i] = f2bf(in[r * (size_t)srcStride + c]);
  }
}

// ---------- depthwise causal conv (width 4) + SiLU ; emits f32 u and bf16 u ----------
// xr = x_and_res [NTOK, 2*DI]; xs = cols [0,DI)
__global__ void conv_silu_kernel(const float* __restrict__ xr, const float* __restrict__ cw,
                                 float* __restrict__ u, ushort_t* __restrict__ u_bf) {
  const size_t n = (size_t)NTOK * DI;
  for (size_t i = (size_t)blockIdx.x * blockDim.x + threadIdx.x; i < n;
       i += (size_t)gridDim.x * blockDim.x) {
    const int d      = (int)(i & (DI - 1));
    const size_t row = i >> 11;                 // token index (DI = 2048)
    const int t      = (int)(row & (L_SZ - 1)); // position within sequence
    float acc = 0.0f;
#pragma unroll
    for (int j = 0; j < DC; ++j) {
      const int off = j - (DC - 1);             // tap offset: t-3 .. t
      if (t + off >= 0)
        acc += cw[d * DC + j] * xr[(row + (size_t)(long)off) * (size_t)(2 * DI) + d];
    }
    const float v = acc * sigmoidf_(acc);
    u[i]    = v;
    u_bf[i] = f2bf(v);
  }
}

// ---------- chunked parallel scan ----------
// Thread layout for passes A/C: gid = ((b*NCH + c)*DI + d)*DS + s  (s fastest -> 16-lane groups)
// Pass A: per (b,c,d,s) compute chunk transition product aprod and zero-start local state hloc.
__global__ void scan_chunk_local(const float* __restrict__ u, const float* __restrict__ dtf,
                                 const float* __restrict__ xdbl, const float* __restrict__ A_log,
                                 float* __restrict__ aprod, float* __restrict__ hloc) {
  const int gid = (int)(blockIdx.x * blockDim.x + threadIdx.x);
  if (gid >= B_SZ * NCH * DI * DS) return;
  const int s = gid & (DS - 1);
  const int d = (gid >> 4) & (DI - 1);
  const int c = (gid >> 15) & (NCH - 1);        // DI*DS = 2^15
  const int b = gid >> 18;                      // NCH*DI*DS = 2^18
  const float Av = -expf(A_log[d * DS + s]);
  float ap = 1.0f, h = 0.0f;
  const size_t row0 = (size_t)b * L_SZ + (size_t)c * CHL;
  for (int t = 0; t < CHL; ++t) {
    const size_t row = row0 + t;
    const float dtv = dtf[row * DI + d];
    const float du  = dtv * u[row * DI + d];
    const float dA  = expf(dtv * Av);
    h  = dA * h + du * xdbl[row * 96 + DTR + s];
    ap *= dA;
  }
  aprod[gid] = ap;
  hloc[gid]  = h;
}

// Pass B: tiny serial prefix over the NCH chunks per (b,d,s) -> start state of each chunk.
__global__ void scan_chunk_prefix(const float* __restrict__ aprod, const float* __restrict__ hloc,
                                  float* __restrict__ hstart) {
  const int gid = (int)(blockIdx.x * blockDim.x + threadIdx.x);
  if (gid >= B_SZ * DI * DS) return;
  const int s = gid & (DS - 1);
  const int d = (gid >> 4) & (DI - 1);
  const int b = gid >> 15;
  float h = 0.0f;
#pragma unroll
  for (int c = 0; c < NCH; ++c) {
    const size_t idx = (((size_t)b * NCH + c) * DI + d) * DS + s;
    hstart[idx] = h;
    h = aprod[idx] * h + hloc[idx];
  }
}

// Pass C: replay chunk from hstart, y_t = sum_s h_s * C_s via 16-lane xor-shuffle reduction.
__global__ void scan_chunk_emit(const float* __restrict__ u, const float* __restrict__ dtf,
                                const float* __restrict__ xdbl, const float* __restrict__ A_log,
                                const float* __restrict__ hstart, float* __restrict__ y) {
  const int gid = (int)(blockIdx.x * blockDim.x + threadIdx.x);
  if (gid >= B_SZ * NCH * DI * DS) return;
  const int s = gid & (DS - 1);
  const int d = (gid >> 4) & (DI - 1);
  const int c = (gid >> 15) & (NCH - 1);
  const int b = gid >> 18;
  const float Av = -expf(A_log[d * DS + s]);
  float h = hstart[gid];
  const size_t row0 = (size_t)b * L_SZ + (size_t)c * CHL;
  for (int t = 0; t < CHL; ++t) {
    const size_t row = row0 + t;
    const float dtv = dtf[row * DI + d];
    const float du  = dtv * u[row * DI + d];
    const float dA  = expf(dtv * Av);
    h = dA * h + du * xdbl[row * 96 + DTR + s];
    float yc = h * xdbl[row * 96 + DTR + DS + s];
    yc += __shfl_xor(yc, 8, 16);
    yc += __shfl_xor(yc, 4, 16);
    yc += __shfl_xor(yc, 2, 16);
    yc += __shfl_xor(yc, 1, 16);
    if (s == 0) y[row * DI + d] = yc;
  }
}

// ---------- gating: g = (y + u*D) * silu(res), output bf16 for final GEMM ----------
__global__ void gate_kernel(const float* __restrict__ y, const float* __restrict__ u,
                            const float* __restrict__ xr, const float* __restrict__ Dp,
                            ushort_t* __restrict__ g) {
  const size_t n = (size_t)NTOK * DI;
  for (size_t i = (size_t)blockIdx.x * blockDim.x + threadIdx.x; i < n;
       i += (size_t)gridDim.x * blockDim.x) {
    const int d = (int)(i & (DI - 1));
    const size_t row = i >> 11;
    const float res = xr[row * (size_t)(2 * DI) + DI + d];
    const float v = (y[i] + u[i] * Dp[d]) * (res * sigmoidf_(res));
    g[i] = f2bf(v);
  }
}

static inline int gemm_blocks(int M, int N, int nblk) {
  long waves = (long)(M / 16) * (long)(N / (16 * nblk));
  return (int)((waves * 32 + 255) / 256);
}

extern "C" void kernel_launch(void* const* d_in, const int* in_sizes, int n_in,
                              void* d_out, int out_size, void* d_ws, size_t ws_size,
                              hipStream_t stream) {
  (void)in_sizes; (void)n_in; (void)out_size; (void)ws_size;
  const float* x       = (const float*)d_in[0];
  const float* W_in    = (const float*)d_in[1];
  const float* conv_w  = (const float*)d_in[2];
  const float* W_x     = (const float*)d_in[3];
  const float* W_dt    = (const float*)d_in[4];
  const float* b_dt    = (const float*)d_in[5];
  const float* A_log   = (const float*)d_in[6];
  const float* D_param = (const float*)d_in[7];
  const float* W_out   = (const float*)d_in[8];
  float* out = (float*)d_out;

  // ---- carve workspace (~125 MB) ----
  char* base = (char*)d_ws;
  size_t off = 0;
  auto alloc = [&](size_t bytes) -> void* {
    off = (off + 255) & ~(size_t)255;
    void* p = base + off;
    off += bytes;
    return p;
  };
  float*    xr     = (float*)   alloc((size_t)NTOK * 2 * DI * 4);      // x_and_res
  float*    uu     = (float*)   alloc((size_t)NTOK * DI * 4);          // u (f32)
  float*    xdbl   = (float*)   alloc((size_t)NTOK * 96 * 4);          // [dt_raw|B|C]
  float*    dtf    = (float*)   alloc((size_t)NTOK * DI * 4);          // softplus dt
  float*    yy     = (float*)   alloc((size_t)NTOK * DI * 4);          // scan output
  float*    aprod  = (float*)   alloc((size_t)B_SZ * NCH * DI * DS * 4);
  float*    hloc   = (float*)   alloc((size_t)B_SZ * NCH * DI * DS * 4);
  float*    hstart = (float*)   alloc((size_t)B_SZ * NCH * DI * DS * 4);
  ushort_t* x_bf   = (ushort_t*)alloc((size_t)NTOK * DM * 2);
  ushort_t* u_bf   = (ushort_t*)alloc((size_t)NTOK * DI * 2);
  ushort_t* dtr_bf = (ushort_t*)alloc((size_t)NTOK * DTR * 2);
  ushort_t* g_bf   = (ushort_t*)alloc((size_t)NTOK * DI * 2);
  ushort_t* WinT   = (ushort_t*)alloc((size_t)(2 * DI) * DM * 2);      // [4096,1024]
  ushort_t* WxT    = (ushort_t*)alloc((size_t)96 * DI * 2);            // [96,2048]
  ushort_t* WdtT   = (ushort_t*)alloc((size_t)DI * DTR * 2);           // [2048,64]
  ushort_t* WoutT  = (ushort_t*)alloc((size_t)DM * DI * 2);            // [1024,2048]

  // ---- weight transposes to bf16 [N,K] ----
  dim3 tb(32, 8);
  transpose_f32_to_bf16<<<dim3(4096 / 32, 1024 / 32), tb, 0, stream>>>(W_in,  WinT,  1024, 4096);
  transpose_f32_to_bf16<<<dim3(3,          2048 / 32), tb, 0, stream>>>(W_x,   WxT,   2048, 96);
  transpose_f32_to_bf16<<<dim3(2048 / 32,  2),         tb, 0, stream>>>(W_dt,  WdtT,  64,   2048);
  transpose_f32_to_bf16<<<dim3(1024 / 32,  2048 / 32), tb, 0, stream>>>(W_out, WoutT, 2048, 1024);

  // 1) x_and_res = x @ W_in           (M=2048, N=4096, K=1024)
  convert_bf16_kernel<<<2048, 256, 0, stream>>>(x, DM, x_bf, NTOK, DM);
  wmma_gemm_bf16_kernel<4, 0><<<gemm_blocks(NTOK, 2 * DI, 4), 256, 0, stream>>>(
      x_bf, DM, WinT, DM, xr, 2 * DI, NTOK, 2 * DI, DM, nullptr);

  // 2) u = silu(causal depthwise conv(xs))   (also emits bf16 copy)
  conv_silu_kernel<<<2048, 256, 0, stream>>>(xr, conv_w, uu, u_bf);

  // 3) x_dbl = u @ W_x                (M=2048, N=96, K=2048) -> NBLK=2 (96 = 3*32)
  wmma_gemm_bf16_kernel<2, 0><<<gemm_blocks(NTOK, 96, 2), 256, 0, stream>>>(
      u_bf, DI, WxT, DI, xdbl, 96, NTOK, 96, DI, nullptr);

  // 4) dt = softplus(dt_raw @ W_dt + b_dt)   (M=2048, N=2048, K=64)
  convert_bf16_kernel<<<512, 256, 0, stream>>>(xdbl, 96, dtr_bf, NTOK, DTR);
  wmma_gemm_bf16_kernel<4, 1><<<gemm_blocks(NTOK, DI, 4), 256, 0, stream>>>(
      dtr_bf, DTR, WdtT, DTR, dtf, DI, NTOK, DI, DTR, b_dt);

  // 5) chunked parallel selective scan
  {
    const int nA = B_SZ * NCH * DI * DS;        // 524288 threads
    scan_chunk_local<<<nA / 256, 256, 0, stream>>>(uu, dtf, xdbl, A_log, aprod, hloc);
    const int nB = B_SZ * DI * DS;              // 65536 threads
    scan_chunk_prefix<<<nB / 256, 256, 0, stream>>>(aprod, hloc, hstart);
    scan_chunk_emit<<<nA / 256, 256, 0, stream>>>(uu, dtf, xdbl, A_log, hstart, yy);
  }

  // 6) gate + final projection        (M=2048, N=1024, K=2048)
  gate_kernel<<<2048, 256, 0, stream>>>(yy, uu, xr, D_param, g_bf);
  wmma_gemm_bf16_kernel<4, 0><<<gemm_blocks(NTOK, DM, 4), 256, 0, stream>>>(
      g_bf, DI, WoutT, DI, out, DM, NTOK, DM, DI, nullptr);
}
